// SampleAndGroup_9165460210299
// MI455X (gfx1250) — matrix operand order repaired and checked by reading, not compiled, not run on hardware
//
#include <hip/hip_runtime.h>

typedef __attribute__((ext_vector_type(16))) _Float16 v16h;
typedef __attribute__((ext_vector_type(8)))  float    v8f;

#define BATCH 16
#define NPTS  4096
#define DIM   64
#define NP1   512
#define NS1   32
#define NP2   128
#define NS2   16
#define BN_EPS 1e-5f

// ---------------------------------------------------------------- center ----
__global__ void __launch_bounds__(256)
center_kernel(const float* __restrict__ xyz, float* __restrict__ center_ws,
              float* __restrict__ center_out) {
  int b = blockIdx.x, t = threadIdx.x;
  float sx = 0.f, sy = 0.f, sz = 0.f;
  for (int n = t; n < NPTS; n += 256) {
    const float* p = xyz + ((size_t)b * NPTS + n) * 3;
    sx += p[0]; sy += p[1]; sz += p[2];
  }
  __shared__ float sm[3][256];
  sm[0][t] = sx; sm[1][t] = sy; sm[2][t] = sz;
  __syncthreads();
  for (int off = 128; off > 0; off >>= 1) {
    if (t < off) {
      sm[0][t] += sm[0][t + off];
      sm[1][t] += sm[1][t + off];
      sm[2][t] += sm[2][t + off];
    }
    __syncthreads();
  }
  if (t < 3) {
    float c = sm[t][0] * (1.0f / NPTS);
    center_ws[b * 3 + t]  = c;
    center_out[b * 3 + t] = c;
  }
}

// ------------------------------------------------- conv1 (6->64) + pts3 ----
__global__ void __launch_bounds__(256)
conv1_kernel(const float* __restrict__ xyz, const float* __restrict__ f,
             const float* __restrict__ center, const float* __restrict__ w,
             const float* __restrict__ g, const float* __restrict__ bb,
             const float* __restrict__ m, const float* __restrict__ v,
             float* __restrict__ h, float* __restrict__ pts3) {
  int t = threadIdx.x;
  int sub = t >> 6;                      // 4 points per block
  int o   = t & 63;
  int n = blockIdx.x * 4 + sub, b = blockIdx.y;
  __shared__ float x6[4][6];
  size_t pidx = (size_t)b * NPTS + n;
  if (o < 3)      x6[sub][o] = xyz[pidx * 3 + o] - center[b * 3 + o];
  else if (o < 6) x6[sub][o] = f[pidx * 3 + (o - 3)];
  __syncthreads();
  float acc = 0.f;
#pragma unroll
  for (int d = 0; d < 6; ++d) acc += x6[sub][d] * w[o * 6 + d];
  float scale = g[o] * rsqrtf(v[o] + BN_EPS);
  float val = fmaxf(scale * (acc - m[o]) + bb[o], 0.f);
  h[pidx * DIM + o] = val;
  if (o < 3) pts3[pidx * 3 + o] = x6[sub][o];
}

// ------------------------------------------- wave32 (val,idx) reductions ----
__device__ inline void wave_argmax(float& v, int& i) {
#pragma unroll
  for (int off = 16; off > 0; off >>= 1) {
    float v2 = __shfl_xor(v, off, 32);
    int   i2 = __shfl_xor(i, off, 32);
    if (v2 > v || (v2 == v && i2 < i)) { v = v2; i = i2; }
  }
}
__device__ inline void wave_argmin(float& v, int& i) {
#pragma unroll
  for (int off = 16; off > 0; off >>= 1) {
    float v2 = __shfl_xor(v, off, 32);
    int   i2 = __shfl_xor(i, off, 32);
    if (v2 < v || (v2 == v && i2 < i)) { v = v2; i = i2; }
  }
}

// -------------------------------------------- farthest point sampling ----
__global__ void __launch_bounds__(256)
fps_kernel(const float* __restrict__ pts, int n, int npoint,
           int* __restrict__ idx) {
  int b = blockIdx.x, t = threadIdx.x;
  const int T = 256;
  int lane = t & 31, wid = t >> 5;
  __shared__ float dist[NPTS];
  __shared__ float swv[8];
  __shared__ int   swi[8];
  __shared__ float farp[3];
  __shared__ int   farIdx;
  const float* P = pts + (size_t)b * n * 3;
  for (int i = t; i < n; i += T) dist[i] = 1e10f;
  if (t == 0) farIdx = 0;
  __syncthreads();
  for (int s = 0; s < npoint; ++s) {
    int far = farIdx;
    if (t == 0) idx[b * npoint + s] = far;
    if (t < 3) farp[t] = P[far * 3 + t];
    __syncthreads();
    float fx = farp[0], fy = farp[1], fz = farp[2];
    float bestV = -1.f; int bestI = 0;
    for (int i = t; i < n; i += T) {
      float dx = P[i * 3 + 0] - fx;
      float dy = P[i * 3 + 1] - fy;
      float dz = P[i * 3 + 2] - fz;
      float d  = fminf(dist[i], dx * dx + dy * dy + dz * dz);
      dist[i] = d;
      if (d > bestV) { bestV = d; bestI = i; }
    }
    wave_argmax(bestV, bestI);
    if (lane == 0) { swv[wid] = bestV; swi[wid] = bestI; }
    __syncthreads();
    if (t == 0) {
      float bv = swv[0]; int bi = swi[0];
#pragma unroll
      for (int w2 = 1; w2 < 8; ++w2)
        if (swv[w2] > bv || (swv[w2] == bv && swi[w2] < bi)) { bv = swv[w2]; bi = swi[w2]; }
      farIdx = bi;
    }
    __syncthreads();
  }
}

// ------------------------------------------------------- gather xyz(3) ----
__global__ void __launch_bounds__(64)
gather3_kernel(const float* __restrict__ pts, const int* __restrict__ idx,
               int n, int npoint, float* __restrict__ out) {
  int b = blockIdx.y;
  int s = blockIdx.x * 64 + threadIdx.x;
  if (s < npoint) {
    int i = idx[b * npoint + s];
#pragma unroll
    for (int c = 0; c < 3; ++c)
      out[((size_t)b * npoint + s) * 3 + c] = pts[((size_t)b * n + i) * 3 + c];
  }
}

// ----------------------------------------------------- kNN (iterated) ----
__global__ void __launch_bounds__(256)
knn_kernel(const float* __restrict__ q, const float* __restrict__ pts,
           int n, int npoint, int nsample, int* __restrict__ knn) {
  int b = blockIdx.y, s = blockIdx.x, t = threadIdx.x;
  const int T = 256;
  int lane = t & 31, wid = t >> 5;
  __shared__ float d[NPTS];
  __shared__ float swv[8];
  __shared__ int   swi[8];
  const float* P = pts + (size_t)b * n * 3;
  float qx = q[((size_t)b * npoint + s) * 3 + 0];
  float qy = q[((size_t)b * npoint + s) * 3 + 1];
  float qz = q[((size_t)b * npoint + s) * 3 + 2];
  for (int i = t; i < n; i += T) {
    float dx = P[i * 3 + 0] - qx;
    float dy = P[i * 3 + 1] - qy;
    float dz = P[i * 3 + 2] - qz;
    d[i] = dx * dx + dy * dy + dz * dz;
  }
  __syncthreads();
  for (int k = 0; k < nsample; ++k) {
    float bv = 1e30f; int bi = 0;
    for (int i = t; i < n; i += T) {
      float vv = d[i];
      if (vv < bv) { bv = vv; bi = i; }
    }
    wave_argmin(bv, bi);
    if (lane == 0) { swv[wid] = bv; swi[wid] = bi; }
    __syncthreads();
    if (t == 0) {
      float mv = swv[0]; int mi = swi[0];
#pragma unroll
      for (int w2 = 1; w2 < 8; ++w2)
        if (swv[w2] < mv || (swv[w2] == mv && swi[w2] < mi)) { mv = swv[w2]; mi = swi[w2]; }
      knn[((size_t)b * npoint + s) * nsample + k] = mi;
      d[mi] = 1e30f;
    }
    __syncthreads();
  }
}

// ------------- weights f32 (o=64, d=128) -> f16 WMMA-fragment order --------
// wf[((nt*4+ks)*32 + lane)*16 + i] = B[k][col],  col = nt*16 + (lane&15),
// k = ks*32 + (lane>>4)*16 + i.  Each lane's fragment = 32 contiguous bytes.
__global__ void __launch_bounds__(256)
wfrag_kernel(const float* __restrict__ w, _Float16* __restrict__ wf) {
  int tid = blockIdx.x * 256 + threadIdx.x;
  if (tid < 16 * 32 * 16) {
    int i  = tid & 15;
    int L  = (tid >> 4) & 31;
    int ks = (tid >> 9) & 3;
    int nt = tid >> 11;
    int col = nt * 16 + (L & 15);
    int k   = ks * 32 + (L >> 4) * 16 + i;
    wf[tid] = (_Float16)w[col * 128 + k];
  }
}

// ---------------- local op: gather + center + GEMM(WMMA) + BN/ReLU/max ----
// One wave per group.  A tile (NSAMPLE x 128 f16) in LDS, padded row stride
// 136 halves (68 dwords) -> bank-conflict-free ds_load_b128.  B fragments
// loaded straight from global in fragment order (L2-resident, coalesced).
template <int NSAMPLE>
__global__ void __launch_bounds__(32)
local_op_kernel(const float* __restrict__ feat, const int* __restrict__ fpsIdx,
                const int* __restrict__ knn, const _Float16* __restrict__ wf,
                const float* __restrict__ g, const float* __restrict__ bb,
                const float* __restrict__ m, const float* __restrict__ v,
                int n, int npoint, float* __restrict__ out) {
  constexpr int MT   = NSAMPLE / 16;   // M tiles (2 or 1)
  constexpr int AROW = 68;             // dwords per padded A row (136 halves)
  int b = blockIdx.y, s = blockIdx.x;
  int L = threadIdx.x;                 // 0..31 (wave32)
  __shared__ __align__(16) unsigned int Au[NSAMPLE * AROW];
  __shared__ float np_s[DIM];
  const _Float16* Ah = (const _Float16*)Au;

  // center point features
  int ci = fpsIdx[b * npoint + s];
  const float* cp = feat + ((size_t)b * n + ci) * DIM;
  for (int o = L; o < DIM; o += 32) np_s[o] = cp[o];
  __syncthreads();

  // gather + center neighbor features -> A tile (packed 32-bit LDS stores)
  if (L < NSAMPLE) {
    int gi = knn[((size_t)b * npoint + s) * NSAMPLE + L];
    const float* gp = feat + ((size_t)b * n + gi) * DIM;
    unsigned int* row = Au + L * AROW;
#pragma unroll
    for (int o = 0; o < DIM; o += 2) {
      union { _Float16 h[2]; unsigned int u; } pc, pn;
      pc.h[0] = (_Float16)(gp[o]     - np_s[o]);
      pc.h[1] = (_Float16)(gp[o + 1] - np_s[o + 1]);
      pn.h[0] = (_Float16)np_s[o];
      pn.h[1] = (_Float16)np_s[o + 1];
      row[o >> 1]        = pc.u;       // centered   -> cols 0..63
      row[32 + (o >> 1)] = pn.u;       // new_pts    -> cols 64..127
    }
  }
  __syncthreads();

  const int lm = L & 15, hi = L >> 4;
  const v16h* wfv = (const v16h*)wf;

#pragma unroll
  for (int nt = 0; nt < 4; ++nt) {     // 4 tiles of 16 output channels
    v8f acc[MT];
#pragma unroll
    for (int mt = 0; mt < MT; ++mt) acc[mt] = (v8f){};
    const int col = nt * 16 + lm;
#pragma unroll
    for (int ks = 0; ks < 4; ++ks) {   // K = 128 in steps of 32
      v16h bf = wfv[(nt * 4 + ks) * 32 + L];     // 32B coalesced global load
#pragma unroll
      for (int mt = 0; mt < MT; ++mt) {
        v16h af;
        const int row = mt * 16 + lm;
#pragma unroll
        for (int i = 0; i < 8; ++i) {  // two 16B chunks -> ds_load_b128 x2
          int kk = ks * 32 + hi * 16 + 2 * i;
          af[2 * i]     = Ah[row * (AROW * 2) + kk];
          af[2 * i + 1] = Ah[row * (AROW * 2) + kk + 1];
        }
        acc[mt] = __builtin_amdgcn_wmma_f32_16x16x32_f16(
            false, af, false, bf, (short)0, acc[mt], false, false);
      }
    }
    // BN + ReLU + max over neighborhood.  Lane holds column col, rows
    // hi*8..hi*8+7 per M-tile; shfl_xor(16) merges the two lane halves.
    float scale = g[col] * rsqrtf(v[col] + BN_EPS);
    float bias  = bb[col] - m[col] * scale;
    float mx = 0.f;                    // relu() >= 0 -> 0 is a safe identity
#pragma unroll
    for (int mt = 0; mt < MT; ++mt)
#pragma unroll
      for (int i = 0; i < 8; ++i)
        mx = fmaxf(mx, fmaxf(acc[mt][i] * scale + bias, 0.f));
    mx = fmaxf(mx, __shfl_xor(mx, 16, 32));
    if (L < 16) out[((size_t)b * npoint + s) * DIM + col] = mx;
  }
}

// --------------------------------------------------------------- launch ----
extern "C" void kernel_launch(void* const* d_in, const int* in_sizes, int n_in,
                              void* d_out, int out_size, void* d_ws, size_t ws_size,
                              hipStream_t stream) {
  const float* xyz     = (const float*)d_in[0];
  const float* f       = (const float*)d_in[1];
  const float* conv1_w = (const float*)d_in[2];
  const float* bn1_g   = (const float*)d_in[3];
  const float* bn1_b   = (const float*)d_in[4];
  const float* bn1_m   = (const float*)d_in[5];
  const float* bn1_v   = (const float*)d_in[6];
  const float* w0      = (const float*)d_in[7];
  const float* g0      = (const float*)d_in[8];
  const float* b0      = (const float*)d_in[9];
  const float* m0      = (const float*)d_in[10];
  const float* v0      = (const float*)d_in[11];
  const float* w1      = (const float*)d_in[12];
  const float* g1      = (const float*)d_in[13];
  const float* b1      = (const float*)d_in[14];
  const float* m1      = (const float*)d_in[15];
  const float* v1      = (const float*)d_in[16];

  float* outp = (float*)d_out;         // [0..47] center, [48..] feat1

  // workspace layout (64B-aligned slices)
  char* ws = (char*)d_ws;
  size_t off = 0;
  auto alloc = [&](size_t bytes) {
    char* p = ws + off;
    off += (bytes + 63) & ~(size_t)63;
    return p;
  };
  float* center  = (float*)alloc(BATCH * 3 * sizeof(float));
  float* pts3    = (float*)alloc((size_t)BATCH * NPTS * 3 * sizeof(float));
  float* h       = (float*)alloc((size_t)BATCH * NPTS * DIM * sizeof(float));
  int*   fps1    = (int*)  alloc((size_t)BATCH * NP1 * sizeof(int));
  float* nxyz1   = (float*)alloc((size_t)BATCH * NP1 * 3 * sizeof(float));
  int*   knn1    = (int*)  alloc((size_t)BATCH * NP1 * NS1 * sizeof(int));
  float* feat0   = (float*)alloc((size_t)BATCH * NP1 * DIM * sizeof(float));
  int*   fps2    = (int*)  alloc((size_t)BATCH * NP2 * sizeof(int));
  float* nxyz2   = (float*)alloc((size_t)BATCH * NP2 * 3 * sizeof(float));
  int*   knn2    = (int*)  alloc((size_t)BATCH * NP2 * NS2 * sizeof(int));
  _Float16* w0f  = (_Float16*)alloc((size_t)128 * DIM * sizeof(_Float16));
  _Float16* w1f  = (_Float16*)alloc((size_t)128 * DIM * sizeof(_Float16));

  // 1. mean center (also writes output[0:48])
  center_kernel<<<BATCH, 256, 0, stream>>>(xyz, center, outp);
  // 2. pointwise conv 6->64 + BN + ReLU, emit centered xyz
  conv1_kernel<<<dim3(NPTS / 4, BATCH), 256, 0, stream>>>(
      xyz, f, center, conv1_w, bn1_g, bn1_b, bn1_m, bn1_v, h, pts3);
  // 3. stage-1 FPS (512 of 4096)
  fps_kernel<<<BATCH, 256, 0, stream>>>(pts3, NPTS, NP1, fps1);
  // 4. gather sampled coordinates
  gather3_kernel<<<dim3((NP1 + 63) / 64, BATCH), 64, 0, stream>>>(
      pts3, fps1, NPTS, NP1, nxyz1);
  // 5. stage-1 kNN (top 32)
  knn_kernel<<<dim3(NP1, BATCH), 256, 0, stream>>>(
      nxyz1, pts3, NPTS, NP1, NS1, knn1);
  // 6. weights -> f16 WMMA-fragment order
  wfrag_kernel<<<(16 * 32 * 16 + 255) / 256, 256, 0, stream>>>(w0, w0f);
  wfrag_kernel<<<(16 * 32 * 16 + 255) / 256, 256, 0, stream>>>(w1, w1f);
  // 7. stage-1 local op (WMMA) -> feat0
  local_op_kernel<NS1><<<dim3(NP1, BATCH), 32, 0, stream>>>(
      h, fps1, knn1, w0f, g0, b0, m0, v0, NPTS, NP1, feat0);
  // 8. stage-2 FPS (128 of 512)
  fps_kernel<<<BATCH, 256, 0, stream>>>(nxyz1, NP1, NP2, fps2);
  // 9. gather stage-2 coordinates
  gather3_kernel<<<dim3((NP2 + 63) / 64, BATCH), 64, 0, stream>>>(
      nxyz1, fps2, NP1, NP2, nxyz2);
  // 10. stage-2 kNN (top 16)
  knn_kernel<<<dim3(NP2, BATCH), 256, 0, stream>>>(
      nxyz2, nxyz1, NP1, NP2, NS2, knn2);
  // 11. stage-2 local op (WMMA) -> output feat1
  local_op_kernel<NS2><<<dim3(NP2, BATCH), 32, 0, stream>>>(
      feat0, fps2, knn2, w1f, g1, b1, m1, v1, NP1, NP2, outp + BATCH * 3);
}
